// BartSentenceAttention_68212670595383
// MI455X (gfx1250) — matrix operand order repaired and checked by reading, not compiled, not run on hardware
//
#include <hip/hip_runtime.h>
#include <hip/hip_bf16.h>
#include <math.h>

// ---------------------------------------------------------------------------
// BartSentenceAttention for MI455X (gfx1250), wave32 + v_wmma_f32_16x16x32_bf16.
// Roofline: ~512MB unavoidable HBM traffic (~22us @ 23.3TB/s); 39 GFLOP of
// GEMM work fits under that roof only on the 16-bit WMMA path, so all GEMMs
// run bf16 inputs with f32 accumulation, with hidden_states read exactly once
// by the fully fused main kernel. sched_group_barrier hints pin the
// loads->convert->WMMA pipeline so waits are partial instead of full drains.
// ---------------------------------------------------------------------------

typedef __bf16 bf16_t;
typedef __attribute__((ext_vector_type(16))) __bf16 v16bf;
typedef __attribute__((ext_vector_type(8)))  float  v8f;

static constexpr int Bc = 8;          // batch
static constexpr int Sc = 8192;       // sequence
static constexpr int Ec = 1024;       // embed
static constexpr int Dc = 128;        // sentence dim == MAX_SEN
static constexpr int LDP = 133;       // LDS row stride (floats), conflict-free
static constexpr int WPB = 4;         // waves per block in fused kernel

// Scheduling-group hints (soft): 0x002 VALU, 0x008 WMMA/MFMA, 0x020 VMEM read,
// 0x040 VMEM write, 0x100 DS read.
#if defined(__has_builtin)
#if __has_builtin(__builtin_amdgcn_sched_group_barrier)
#define SCHED_GROUP(mask, size, id) __builtin_amdgcn_sched_group_barrier(mask, size, id)
#endif
#endif
#ifndef SCHED_GROUP
#define SCHED_GROUP(mask, size, id)
#endif

__device__ inline v8f v8f_zero() {
  v8f z;
#pragma unroll
  for (int i = 0; i < 8; ++i) z[i] = 0.0f;
  return z;
}

__device__ inline bf16_t to_bf16(float f) { return (bf16_t)f; }

__device__ inline v8f wmma_bf16(v16bf a, v16bf b, v8f c) {
  // D(f32,16x16) = A(bf16,16x32) * B(bf16,32x16) + C
  return __builtin_amdgcn_wmma_f32_16x16x32_bf16(
      /*neg_a=*/false, a, /*neg_b=*/false, b,
      /*c_mod=*/(short)0, c, /*reuse_a=*/false, /*reuse_b=*/false);
}

// Convert 4 float4 fragments (hidden row, K runs [0..7] and [16..23]) into the
// ISA A-operand element order.
__device__ inline v16bf a_from_f4(float4 f0, float4 f1, float4 f2, float4 f3) {
  v16bf a;
  a[0]=to_bf16(f0.x); a[1]=to_bf16(f0.y); a[2]=to_bf16(f0.z); a[3]=to_bf16(f0.w);
  a[4]=to_bf16(f1.x); a[5]=to_bf16(f1.y); a[6]=to_bf16(f1.z); a[7]=to_bf16(f1.w);
  a[8]=to_bf16(f2.x); a[9]=to_bf16(f2.y); a[10]=to_bf16(f2.z); a[11]=to_bf16(f2.w);
  a[12]=to_bf16(f3.x); a[13]=to_bf16(f3.y); a[14]=to_bf16(f3.z); a[15]=to_bf16(f3.w);
  return a;
}

// Read one 16x32 bf16 A tile from an LDS f32 buffer (rows stride LDP).
// ISA A layout: lanes<16 -> K = e (e<8) / e+8 (e>=8); lanes>=16 add +8.
__device__ inline v16bf a_from_lds(const float* sbuf, int k0, int row, int khalf) {
  const float* p = sbuf + row * LDP + k0 + khalf;
  v16bf a;
#pragma unroll
  for (int e = 0; e < 8; ++e)  a[e] = to_bf16(p[e]);
#pragma unroll
  for (int e = 8; e < 16; ++e) a[e] = to_bf16(p[e + 8]);
  return a;
}

// Scatter a 16x128 D-layout accumulator set (8 tiles of v8f) into LDS f32.
__device__ inline void d_to_lds(float* sbuf, const v8f* acc, int nlo, int khalf) {
#pragma unroll
  for (int nt = 0; nt < 8; ++nt)
#pragma unroll
    for (int r = 0; r < 8; ++r)
      sbuf[(khalf + r) * LDP + nt * 16 + nlo] = acc[nt][r];
}

// ---------------------------------------------------------------------------
// K0: per-batch mask scan. Produces clipped cumsum (attention mask bound) and
// the token position of each of the first 128 mask tokens (-1 if absent).
// ---------------------------------------------------------------------------
__global__ __launch_bounds__(1024) void seg_scan_kernel(
    const int* __restrict__ ids, int* __restrict__ csum_clip,
    int* __restrict__ pos) {
  const int b = blockIdx.x;
  const int tid = threadIdx.x;
  const int lane = tid & 31, wid = tid >> 5;
  __shared__ int wsum[32];

  if (tid < 128) pos[b * 128 + tid] = -1;
  __syncthreads();

  const int* idrow = ids + (size_t)b * Sc;
  const int base = tid * 8;   // 1024 threads * 8 = 8192
  int m[8];
  int cnt = 0;
#pragma unroll
  for (int i = 0; i < 8; ++i) { m[i] = (idrow[base + i] == 3); cnt += m[i]; }

  int incl = cnt;
#pragma unroll
  for (int off = 1; off < 32; off <<= 1) {
    int t = __shfl_up(incl, off, 32);
    if (lane >= off) incl += t;
  }
  if (lane == 31) wsum[wid] = incl;
  __syncthreads();
  if (wid == 0) {
    int v = wsum[lane];
    int iv = v;
#pragma unroll
    for (int off = 1; off < 32; off <<= 1) {
      int t = __shfl_up(iv, off, 32);
      if (lane >= off) iv += t;
    }
    wsum[lane] = iv - v;  // exclusive prefix of wave totals
  }
  __syncthreads();

  int c = wsum[wid] + (incl - cnt);  // exclusive count before this chunk
#pragma unroll
  for (int i = 0; i < 8; ++i) {
    if (m[i]) {
      ++c;
      if (c <= 128) pos[b * 128 + (c - 1)] = base + i;
    }
    csum_clip[(size_t)b * Sc + base + i] = (c < 128) ? c : 128;
  }
}

// ---------------------------------------------------------------------------
// Pack an f32 matrix element(k,n) = W[k*ldK + n*ldN] into bf16 WMMA B-operand
// tiles (32x16), tile index = kt*NT + nt, per-lane 16 contiguous bf16 (32B).
// ISA B layout: lanes<16: N=lane, K=e (e<8) / e+8 (e>=8); lanes>=16: N=lane-16, add +8.
// ---------------------------------------------------------------------------
__global__ __launch_bounds__(256) void pack_b_kernel(
    const float* __restrict__ W, bf16_t* __restrict__ out,
    int ldK, int ldN, int KT, int NT,
    size_t wBatchStride, size_t outBatchStride) {
  W   += (size_t)blockIdx.y * wBatchStride;
  out += (size_t)blockIdx.y * outBatchStride;
  const int g = blockIdx.x * blockDim.x + threadIdx.x;
  const int wave = g >> 5;
  const int lane = g & 31;
  if (wave >= KT * NT) return;
  const int nt = wave % NT;
  const int kt = wave / NT;
  const int n  = nt * 16 + (lane & 15);
  const int kb = kt * 32 + ((lane & 16) ? 8 : 0);
  v16bf v;
#pragma unroll
  for (int e = 0; e < 16; ++e)
    v[e] = to_bf16(W[(size_t)(kb + e + (e >= 8 ? 8 : 0)) * ldK + (size_t)n * ldN]);
  *(v16bf*)(out + ((size_t)wave * 32 + lane) * 16) = v;
}

// ---------------------------------------------------------------------------
// K1a: ss[b,m,:] = hidden[b, pos[b,m], :] @ Wv + bv   (zero row if pos<0)
// One wave per 16 sentence rows; K=1024 via bf16 WMMA.
// ---------------------------------------------------------------------------
__global__ __launch_bounds__(32) void sen_pool_wv_kernel(
    const float* __restrict__ hidden, const int* __restrict__ pos,
    const bf16_t* __restrict__ Wv_pk, const float* __restrict__ bv,
    float* __restrict__ ss) {
  const int b = blockIdx.y;
  const int t0 = blockIdx.x * 16;
  const int lane = threadIdx.x & 31;
  const int row = lane & 15;
  const int khalf = (lane & 16) ? 8 : 0;
  const int p = pos[b * 128 + t0 + row];
  const float* aptr = hidden + ((size_t)b * Sc + (size_t)(p >= 0 ? p : 0)) * Ec;

  v8f acc[8];
#pragma unroll
  for (int i = 0; i < 8; ++i) acc[i] = v8f_zero();

  for (int kt = 0; kt < Ec / 32; ++kt) {
    const int k0 = kt * 32 + khalf;
    v16bf a;
    if (p >= 0) {
      float4 f0 = *(const float4*)(aptr + k0);
      float4 f1 = *(const float4*)(aptr + k0 + 4);
      float4 f2 = *(const float4*)(aptr + k0 + 16);
      float4 f3 = *(const float4*)(aptr + k0 + 20);
      a = a_from_f4(f0, f1, f2, f3);
    } else {
#pragma unroll
      for (int e = 0; e < 16; ++e) a[e] = to_bf16(0.0f);
    }
    const bf16_t* bp = Wv_pk + (size_t)kt * 8 * 512 + lane * 16;
    v16bf bt[8];
#pragma unroll
    for (int nt = 0; nt < 8; ++nt) bt[nt] = *(const v16bf*)(bp + nt * 512);
#pragma unroll
    for (int nt = 0; nt < 8; ++nt) acc[nt] = wmma_bf16(a, bt[nt], acc[nt]);
    SCHED_GROUP(0x020, 20, 0);  // group the VMEM reads ahead
    SCHED_GROUP(0x002, 16, 0);  // conversions
    SCHED_GROUP(0x008, 8, 0);   // WMMA burst
  }

  float* srow = ss + (size_t)b * 128 * 128;
  const int nlo = lane & 15;
#pragma unroll
  for (int nt = 0; nt < 8; ++nt) {
    const float bias = bv[nt * 16 + nlo];
#pragma unroll
    for (int r = 0; r < 8; ++r)
      srow[(size_t)(t0 + khalf + r) * 128 + nt * 16 + nlo] = acc[nt][r] + bias;
  }
}

// ---------------------------------------------------------------------------
// K1b: k_sen = ss @ Wk + bk ;  v_sen = ss @ Wvt + bvt   (per batch 128x128)
// ---------------------------------------------------------------------------
__global__ __launch_bounds__(32) void kv_sen_kernel(
    const float* __restrict__ ss,
    const bf16_t* __restrict__ Wk_pk, const bf16_t* __restrict__ Wvt_pk,
    const float* __restrict__ bk, const float* __restrict__ bvt,
    float* __restrict__ ksen, float* __restrict__ vsen) {
  const int b = blockIdx.y;
  const int t0 = blockIdx.x * 16;
  const int lane = threadIdx.x & 31;
  const int row = lane & 15;
  const int khalf = (lane & 16) ? 8 : 0;
  const float* arow = ss + (size_t)b * 128 * 128 + (size_t)(t0 + row) * 128;

  v8f ak[8], av[8];
#pragma unroll
  for (int i = 0; i < 8; ++i) { ak[i] = v8f_zero(); av[i] = v8f_zero(); }

  for (int kc = 0; kc < 4; ++kc) {
    const float* p = arow + kc * 32 + khalf;
    v16bf a;
#pragma unroll
    for (int e = 0; e < 8; ++e)  a[e] = to_bf16(p[e]);
#pragma unroll
    for (int e = 8; e < 16; ++e) a[e] = to_bf16(p[e + 8]);
    const bf16_t* kp = Wk_pk  + (size_t)(kc * 8) * 512 + lane * 16;
    const bf16_t* vp = Wvt_pk + (size_t)(kc * 8) * 512 + lane * 16;
    v16bf bk_t[8], bv_t[8];
#pragma unroll
    for (int nt = 0; nt < 8; ++nt) {
      bk_t[nt] = *(const v16bf*)(kp + nt * 512);
      bv_t[nt] = *(const v16bf*)(vp + nt * 512);
    }
#pragma unroll
    for (int nt = 0; nt < 8; ++nt) {
      ak[nt] = wmma_bf16(a, bk_t[nt], ak[nt]);
      av[nt] = wmma_bf16(a, bv_t[nt], av[nt]);
    }
  }

  const int nlo = lane & 15;
  float* krow = ksen + (size_t)b * 128 * 128;
  float* vrow = vsen + (size_t)b * 128 * 128;
#pragma unroll
  for (int nt = 0; nt < 8; ++nt) {
    const float bkv = bk[nt * 16 + nlo];
    const float bvv = bvt[nt * 16 + nlo];
#pragma unroll
    for (int r = 0; r < 8; ++r) {
      const size_t idx = (size_t)(t0 + khalf + r) * 128 + nt * 16 + nlo;
      krow[idx] = ak[nt][r] + bkv;
      vrow[idx] = av[nt][r] + bvv;
    }
  }
}

// ---------------------------------------------------------------------------
// Fused main kernel: per wave, 16 tokens:
//   q = (hidden@Wq + bq)/sqrt(128)  ->  logits = q@k_senT  -> masked softmax
//   -> attn@v_sen -> @Wo + bo -> out.  hidden read once, out written once.
// LDS (per wave, 16xLDP f32) stages the D-layout -> A-layout transposes;
// per-wave DS ordering makes barriers unnecessary.
// ---------------------------------------------------------------------------
__global__ __launch_bounds__(WPB * 32) void fused_attn_kernel(
    const float* __restrict__ hidden, const int* __restrict__ csum,
    const bf16_t* __restrict__ Wq_pk, const bf16_t* __restrict__ ksenT_pk,
    const bf16_t* __restrict__ vsen_pk, const bf16_t* __restrict__ Wo_pk,
    const float* __restrict__ bq, const float* __restrict__ bo,
    float* __restrict__ out) {
  extern __shared__ float smem[];
  const int lane = threadIdx.x & 31;
  const int wid  = threadIdx.x >> 5;
  const int b    = blockIdx.y;
  const int t0   = (blockIdx.x * WPB + wid) * 16;
  const int row  = lane & 15;
  const int nlo  = row;
  const int khalf = (lane & 16) ? 8 : 0;
  float* sbuf = smem + wid * (16 * LDP);

  // ---- Stage 1: q = (hidden @ Wq + bq) * 128^-0.5 ----
  v8f acc[8];
#pragma unroll
  for (int i = 0; i < 8; ++i) acc[i] = v8f_zero();
  const float* aptr = hidden + ((size_t)b * Sc + (size_t)(t0 + row)) * Ec;
  // Software pipeline: A fragment for step kt is loaded during step kt-1.
  float4 f0 = *(const float4*)(aptr + khalf);
  float4 f1 = *(const float4*)(aptr + khalf + 4);
  float4 f2 = *(const float4*)(aptr + khalf + 16);
  float4 f3 = *(const float4*)(aptr + khalf + 20);
  for (int kt = 0; kt < Ec / 32; ++kt) {
    v16bf a = a_from_f4(f0, f1, f2, f3);
    // Stage all 8 B tiles into registers (one load clause).
    const bf16_t* bp = Wq_pk + (size_t)kt * 8 * 512 + lane * 16;
    v16bf bt[8];
#pragma unroll
    for (int nt = 0; nt < 8; ++nt) bt[nt] = *(const v16bf*)(bp + nt * 512);
    if (kt + 1 < Ec / 32) {   // prefetch next A fragment (stays in flight)
      const int k0 = (kt + 1) * 32 + khalf;
      f0 = *(const float4*)(aptr + k0);
      f1 = *(const float4*)(aptr + k0 + 4);
      f2 = *(const float4*)(aptr + k0 + 16);
      f3 = *(const float4*)(aptr + k0 + 20);
    }
#pragma unroll
    for (int nt = 0; nt < 8; ++nt) acc[nt] = wmma_bf16(a, bt[nt], acc[nt]);
    // Pipeline shape: all VMEM reads (16 B + 4 A-prefetch) first, then the
    // current-A conversions, then the 8-WMMA burst with partial load waits.
    SCHED_GROUP(0x020, 20, 0);
    SCHED_GROUP(0x002, 16, 0);
    SCHED_GROUP(0x008, 8, 0);
  }
  const float qscale = 0.08838834764831845f;  // 1/sqrt(128)
#pragma unroll
  for (int nt = 0; nt < 8; ++nt) {
    const float bias = bq[nt * 16 + nlo];
#pragma unroll
    for (int r = 0; r < 8; ++r) acc[nt][r] = (acc[nt][r] + bias) * qscale;
  }
  d_to_lds(sbuf, acc, nlo, khalf);

  // ---- Stage 2: logits = q @ k_sen^T ----
  v8f lg[8];
#pragma unroll
  for (int i = 0; i < 8; ++i) lg[i] = v8f_zero();
  const bf16_t* kp = ksenT_pk + (size_t)b * (4 * 8 * 512) + lane * 16;
  for (int kc = 0; kc < 4; ++kc) {
    v16bf a = a_from_lds(sbuf, kc * 32, row, khalf);
    v16bf bt[8];
#pragma unroll
    for (int nt = 0; nt < 8; ++nt)
      bt[nt] = *(const v16bf*)(kp + (kc * 8 + nt) * 512);
#pragma unroll
    for (int nt = 0; nt < 8; ++nt) lg[nt] = wmma_bf16(a, bt[nt], lg[nt]);
    SCHED_GROUP(0x100, 8, 0);   // LDS reads for A
    SCHED_GROUP(0x020, 16, 0);  // B-tile loads
    SCHED_GROUP(0x008, 8, 0);   // WMMA burst
  }

  // ---- Stage 3: masked softmax over the 128 sentences ----
  const float NEGMAX = -3.4028234663852886e38f;  // finfo(f32).min, like ref
  int nv[8];
#pragma unroll
  for (int r = 0; r < 8; ++r) nv[r] = csum[(size_t)b * Sc + t0 + khalf + r];
#pragma unroll
  for (int r = 0; r < 8; ++r) {
    float vv[8];
    float mx = NEGMAX;
#pragma unroll
    for (int nt = 0; nt < 8; ++nt) {
      const int c = nt * 16 + nlo;
      const float x = (c < nv[r]) ? lg[nt][r] : NEGMAX;
      vv[nt] = x;
      mx = fmaxf(mx, x);
    }
#pragma unroll
    for (int o = 1; o < 16; o <<= 1) mx = fmaxf(mx, __shfl_xor(mx, o, 32));
    float s = 0.0f;
#pragma unroll
    for (int nt = 0; nt < 8; ++nt) {
      const float e = __expf(vv[nt] - mx);
      vv[nt] = e;
      s += e;
    }
#pragma unroll
    for (int o = 1; o < 16; o <<= 1) s += __shfl_xor(s, o, 32);
    const float inv = 1.0f / s;
#pragma unroll
    for (int nt = 0; nt < 8; ++nt) {
      const int c = nt * 16 + nlo;
      lg[nt][r] = (c < nv[r]) ? vv[nt] * inv : 0.0f;  // ref: softmax * mask
    }
  }

  // ---- Stage 4: o = attn @ v_sen ----
  d_to_lds(sbuf, lg, nlo, khalf);
  v8f ov[8];
#pragma unroll
  for (int i = 0; i < 8; ++i) ov[i] = v8f_zero();
  const bf16_t* vp = vsen_pk + (size_t)b * (4 * 8 * 512) + lane * 16;
  for (int kc = 0; kc < 4; ++kc) {
    v16bf a = a_from_lds(sbuf, kc * 32, row, khalf);
    v16bf bt[8];
#pragma unroll
    for (int nt = 0; nt < 8; ++nt)
      bt[nt] = *(const v16bf*)(vp + (kc * 8 + nt) * 512);
#pragma unroll
    for (int nt = 0; nt < 8; ++nt) ov[nt] = wmma_bf16(a, bt[nt], ov[nt]);
    SCHED_GROUP(0x100, 8, 0);
    SCHED_GROUP(0x020, 16, 0);
    SCHED_GROUP(0x008, 8, 0);
  }

  // ---- Stage 5: out = o @ Wo + bo ----
  d_to_lds(sbuf, ov, nlo, khalf);
  v16bf a4[4];
#pragma unroll
  for (int kc = 0; kc < 4; ++kc) a4[kc] = a_from_lds(sbuf, kc * 32, row, khalf);
  float* orow = out + ((size_t)b * Sc + t0) * (size_t)Ec;
  const bf16_t* wop = Wo_pk + lane * 16;
#pragma unroll 2
  for (int nt = 0; nt < Ec / 16; ++nt) {
    v16bf bt[4];
#pragma unroll
    for (int kc = 0; kc < 4; ++kc)
      bt[kc] = *(const v16bf*)(wop + (size_t)(kc * 64 + nt) * 512);
    v8f o = v8f_zero();
#pragma unroll
    for (int kc = 0; kc < 4; ++kc) o = wmma_bf16(a4[kc], bt[kc], o);
    const float bias = bo[nt * 16 + nlo];
#pragma unroll
    for (int r = 0; r < 8; ++r)
      orow[(size_t)(khalf + r) * Ec + nt * 16 + nlo] = o[r] + bias;
    SCHED_GROUP(0x020, 8, 0);   // B-tile reads
    SCHED_GROUP(0x008, 4, 0);   // WMMAs
    SCHED_GROUP(0x040, 8, 0);   // output stores
  }
}

// ---------------------------------------------------------------------------
extern "C" void kernel_launch(void* const* d_in, const int* in_sizes, int n_in,
                              void* d_out, int out_size, void* d_ws, size_t ws_size,
                              hipStream_t stream) {
  const int*   ids    = (const int*)d_in[0];
  const float* hidden = (const float*)d_in[1];
  const float* Wv     = (const float*)d_in[2];
  const float* bv     = (const float*)d_in[3];
  const float* Wq     = (const float*)d_in[4];
  const float* bq     = (const float*)d_in[5];
  const float* Wk     = (const float*)d_in[6];
  const float* bk     = (const float*)d_in[7];
  const float* Wvt    = (const float*)d_in[8];
  const float* bvt    = (const float*)d_in[9];
  const float* Wo     = (const float*)d_in[10];
  const float* bo     = (const float*)d_in[11];
  float* out = (float*)d_out;

  // Workspace carve (256B aligned). Total ~3.3 MB.
  unsigned char* wp = (unsigned char*)d_ws;
  auto take = [&](size_t bytes) -> void* {
    void* p = wp;
    wp += (bytes + 255) & ~(size_t)255;
    return p;
  };
  int*    csum     = (int*)   take((size_t)Bc * Sc * 4);
  int*    pos      = (int*)   take((size_t)Bc * 128 * 4);
  float*  ss       = (float*) take((size_t)Bc * 128 * 128 * 4);
  float*  ksen     = (float*) take((size_t)Bc * 128 * 128 * 4);
  float*  vsen     = (float*) take((size_t)Bc * 128 * 128 * 4);
  bf16_t* Wq_pk    = (bf16_t*)take((size_t)Ec * Dc * 2);
  bf16_t* Wv_pk    = (bf16_t*)take((size_t)Ec * Dc * 2);
  bf16_t* Wk_pk    = (bf16_t*)take((size_t)Dc * Dc * 2);
  bf16_t* Wvt_pk   = (bf16_t*)take((size_t)Dc * Dc * 2);
  bf16_t* Wo_pk    = (bf16_t*)take((size_t)Dc * Ec * 2);
  bf16_t* ksenT_pk = (bf16_t*)take((size_t)Bc * 128 * 128 * 2);
  bf16_t* vsen_pk  = (bf16_t*)take((size_t)Bc * 128 * 128 * 2);

  // K0: mask scan per batch.
  seg_scan_kernel<<<Bc, 1024, 0, stream>>>(ids, csum, pos);

  // Weight packing: grid.x = ceil(tiles*32/256).
  auto pgrid = [](int KT, int NT) { return dim3((KT * NT * 32 + 255) / 256, 1); };
  pack_b_kernel<<<pgrid(32, 8),  256, 0, stream>>>(Wq,  Wq_pk,  Dc, 1, 32, 8,  0, 0);
  pack_b_kernel<<<pgrid(32, 8),  256, 0, stream>>>(Wv,  Wv_pk,  Dc, 1, 32, 8,  0, 0);
  pack_b_kernel<<<pgrid(4, 8),   256, 0, stream>>>(Wk,  Wk_pk,  Dc, 1, 4,  8,  0, 0);
  pack_b_kernel<<<pgrid(4, 8),   256, 0, stream>>>(Wvt, Wvt_pk, Dc, 1, 4,  8,  0, 0);
  pack_b_kernel<<<pgrid(4, 64),  256, 0, stream>>>(Wo,  Wo_pk,  Ec, 1, 4,  64, 0, 0);

  // K1a: gathered sentence states @ Wv.
  sen_pool_wv_kernel<<<dim3(8, Bc), 32, 0, stream>>>(hidden, pos, Wv_pk, bv, ss);
  // K1b: k_sen / v_sen.
  kv_sen_kernel<<<dim3(8, Bc), 32, 0, stream>>>(ss, Wk_pk, Wvt_pk, bk, bvt, ksen, vsen);

  // Pack per-batch k_sen^T (K=dim,N=sen) and v_sen (K=sen,N=dim).
  pack_b_kernel<<<dim3(4, Bc), 256, 0, stream>>>(ksen, ksenT_pk, 1, 128, 4, 8,
                                                 (size_t)128 * 128, (size_t)4 * 8 * 512);
  pack_b_kernel<<<dim3(4, Bc), 256, 0, stream>>>(vsen, vsen_pk, 128, 1, 4, 8,
                                                 (size_t)128 * 128, (size_t)4 * 8 * 512);

  // Fused main kernel: S/16 row tiles, WPB waves per block.
  const size_t ldsBytes = (size_t)WPB * 16 * LDP * sizeof(float);  // ~34 KB
  fused_attn_kernel<<<dim3(Sc / 16 / WPB, Bc), WPB * 32, ldsBytes, stream>>>(
      hidden, csum, Wq_pk, ksenT_pk, vsen_pk, Wo_pk, bq, bo, out);
}
